// Multi_Head_Self_attention_42734924595675
// MI455X (gfx1250) — matrix operand order repaired and checked by reading, not compiled, not run on hardware
//
#include <hip/hip_runtime.h>
#include <hip/hip_bf16.h>

typedef _Float16 f16_t;
typedef __attribute__((ext_vector_type(16))) _Float16 v16h;
typedef __attribute__((ext_vector_type(4)))  _Float16 v4h;
typedef __attribute__((ext_vector_type(8)))  float    v8f;
typedef unsigned int v4u __attribute__((ext_vector_type(4)));

#define D_MODEL 1024
#define NUM_HEADS 16
#define D_K 64
#define SEQ 2048
#define BATCH 4
#define M_TOTAL (BATCH * SEQ)   // 8192

union Frag16 { v16h h; v4u u[2]; };

__device__ __forceinline__ v16h load_frag(const f16_t* lds, int byte0, int byte1) {
    Frag16 f;
    f.u[0] = *reinterpret_cast<const v4u*>(reinterpret_cast<const char*>(lds) + byte0);
    f.u[1] = *reinterpret_cast<const v4u*>(reinterpret_cast<const char*>(lds) + byte1);
    return f.h;
}

// ---------------------------------------------------------------------------
// f32 -> f16 conversion, 4 elements / thread
// ---------------------------------------------------------------------------
__global__ __launch_bounds__(256) void cvt_f32_f16(const float* __restrict__ in,
                                                   f16_t* __restrict__ out, int n4) {
    int i = blockIdx.x * blockDim.x + threadIdx.x;
    if (i < n4) {
        float4 v = reinterpret_cast<const float4*>(in)[i];
        v4h o = { (_Float16)v.x, (_Float16)v.y, (_Float16)v.z, (_Float16)v.w };
        reinterpret_cast<v4h*>(out)[i] = o;
    }
}

// ---------------------------------------------------------------------------
// NT GEMM: C[m,n] = sum_k A[m,k] * Bw[n,k]   (A: MxK f16, Bw: NxK f16)
// mode==1: write f16 to [B, H, S, 64] layout, optional fused RoPE
// mode==2: write f32 row-major [M, N]
// Block: 256 threads = 8 waves; tile 128(M) x 128(N); K-step 32.
// Wave grid 4(M) x 2(N); each wave owns 32x64 = 2x4 WMMA accumulators.
// Double-buffered LDS: one barrier per K-step.
// ---------------------------------------------------------------------------
__global__ __launch_bounds__(256)
void gemm_nt_kernel(const f16_t* __restrict__ A, const f16_t* __restrict__ Bw,
                    const int* __restrict__ tokpos,
                    f16_t* __restrict__ outH, float* __restrict__ outF,
                    int K, int N, int ropeFlag, int mode)
{
    __shared__ f16_t As[2 * 128 * 40];   // row stride 40 halves = 80 B
    __shared__ f16_t Bs[2 * 128 * 40];

    const int tid   = threadIdx.x;
    const int lane  = tid & 31;
    const int wave  = tid >> 5;
    const int g     = (lane >> 4) & 1;
    const int r     = lane & 15;
    const int waveM = wave >> 1;           // 0..3 -> 32 rows each
    const int waveN = wave & 1;            // 0..1 -> 64 cols each
    const int blockN = blockIdx.x * 128;
    const int blockM = blockIdx.y * 128;

    const int aRow = tid >> 2, aC4 = tid & 3;   // with +64-row second chunk

    v8f acc[2][4] = {};
    v4u aReg[2], bReg[2];

    auto loadGlobal = [&](int k0) {
        #pragma unroll
        for (int i = 0; i < 2; ++i) {
            int row = aRow + i * 64;
            aReg[i] = *reinterpret_cast<const v4u*>(A + (size_t)(blockM + row) * K + k0 + aC4 * 8);
            bReg[i] = *reinterpret_cast<const v4u*>(Bw + (size_t)(blockN + row) * K + k0 + aC4 * 8);
        }
    };
    auto storeLds = [&](int buf) {
        char* ab = reinterpret_cast<char*>(As) + buf * 10240;
        char* bb = reinterpret_cast<char*>(Bs) + buf * 10240;
        #pragma unroll
        for (int i = 0; i < 2; ++i) {
            int row = aRow + i * 64;
            *reinterpret_cast<v4u*>(ab + row * 80 + aC4 * 16) = aReg[i];
            *reinterpret_cast<v4u*>(bb + row * 80 + aC4 * 16) = bReg[i];
        }
    };

    loadGlobal(0);
    storeLds(0);
    __syncthreads();

    int cur = 0;
    for (int k0 = 0; k0 < K; k0 += 32) {
        const bool more = (k0 + 32) < K;
        if (more) loadGlobal(k0 + 32);
        if (k0 + 64 < K) {   // gfx1250 global_prefetch_b8, two K-steps ahead
            __builtin_prefetch((const void*)(A + (size_t)(blockM + aRow) * K + k0 + 64), 0, 1);
            __builtin_prefetch((const void*)(Bw + (size_t)(blockN + aRow) * K + k0 + 64), 0, 1);
        }

        const f16_t* Ac = As + cur * 5120;
        const f16_t* Bc = Bs + cur * 5120;
        v16h af[2], bf[4];
        #pragma unroll
        for (int mt = 0; mt < 2; ++mt) {
            int mrow = waveM * 32 + mt * 16 + r;
            af[mt] = load_frag(Ac, mrow * 80 + 16 * g, mrow * 80 + 32 + 16 * g);
        }
        #pragma unroll
        for (int nt = 0; nt < 4; ++nt) {
            int nrow = waveN * 64 + nt * 16 + r;
            bf[nt] = load_frag(Bc, nrow * 80 + 16 * g, nrow * 80 + 32 + 16 * g);
        }
        #pragma unroll
        for (int mt = 0; mt < 2; ++mt)
            #pragma unroll
            for (int nt = 0; nt < 4; ++nt)
                acc[mt][nt] = __builtin_amdgcn_wmma_f32_16x16x32_f16(
                    false, af[mt], false, bf[nt], (short)0, acc[mt][nt], false, false);

        if (more) storeLds(cur ^ 1);
        __syncthreads();
        cur ^= 1;
    }

    // ---- epilogue: optional RoPE, scatter ----
    #pragma unroll
    for (int mt = 0; mt < 2; ++mt) {
        #pragma unroll
        for (int nt = 0; nt < 4; ++nt) {
            #pragma unroll
            for (int f = 0; f < 8; ++f) {
                int m = blockM + waveM * 32 + mt * 16 + 8 * g + f;
                int n = blockN + waveN * 64 + nt * 16 + r;
                float v = acc[mt][nt][f];
                if (ropeFlag) {
                    float partner = __shfl_xor(v, 1, 32);
                    int d = n & (D_K - 1);
                    // inv_freq = 10000^(-(d&~1)/64) ; ln(10000)=9.2103403719
                    float invf = __expf(-9.210340371976184f * (float)(d & ~1) * (1.0f / 64.0f));
                    float ang = (float)tokpos[m] * invf;
                    float sn, cs;
                    __sincosf(ang, &sn, &cs);
                    v = ((n & 1) == 0) ? (v * cs - partner * sn)
                                       : (partner * sn + v * cs);
                }
                if (mode == 1) {
                    int b = m >> 11, s = m & (SEQ - 1);
                    int h = n >> 6, d = n & (D_K - 1);
                    outH[(((size_t)b * NUM_HEADS + h) * SEQ + s) * D_K + d] = (f16_t)v;
                } else {
                    outF[(size_t)m * N + n] = v;
                }
            }
        }
    }
}

// ---------------------------------------------------------------------------
// Flash-attention (causal, online softmax).
// Q,K,V: [B,H,S,64] f16.  out: [B,S,1024] f16.
// Block: 256 threads = 8 waves; 128 q-rows per block (16 per wave).
// ---------------------------------------------------------------------------
__global__ __launch_bounds__(256)
void attn_kernel(const f16_t* __restrict__ Q, const f16_t* __restrict__ Kt,
                 const f16_t* __restrict__ V, f16_t* __restrict__ outH)
{
    __shared__ f16_t Qs [128 * 72];     // row stride 72 halves = 144 B
    __shared__ f16_t Ks [64 * 72];
    __shared__ f16_t Vts[64 * 72];      // V transposed: [d][kv]
    __shared__ f16_t Ps [8 * 16 * 72];  // per-wave P staging (C-layout -> A-layout)

    const int tid  = threadIdx.x;
    const int lane = tid & 31;
    const int wave = tid >> 5;          // 0..7
    const int g    = (lane >> 4) & 1;
    const int r    = lane & 15;

    const int idx = blockIdx.x;
    const int qt  = idx & 15;           // S/128 = 16 q-tiles
    const int bh  = idx >> 4;           // 0..63
    const int b   = bh >> 4, h = bh & 15;
    const size_t headOff = (size_t)bh * SEQ * D_K;
    const int qBase = qt * 128;
    const int waveRow0 = qBase + wave * 16;   // first global q-row of this wave

    // ---- stage Q tile: 128 rows x 64 halves = 1024 x 16B ----
    #pragma unroll
    for (int i = 0; i < 4; ++i) {
        int u = tid + i * 256;
        int row = u >> 3, c8 = u & 7;
        v4u d = *reinterpret_cast<const v4u*>(Q + headOff + (size_t)(qBase + row) * D_K + c8 * 8);
        *reinterpret_cast<v4u*>(reinterpret_cast<char*>(Qs) + row * 144 + c8 * 16) = d;
    }
    __syncthreads();

    v16h aq[2];
    {
        int row = wave * 16 + r;
        aq[0] = load_frag(Qs, row * 144 + 16 * g,      row * 144 + 32 + 16 * g);
        aq[1] = load_frag(Qs, row * 144 + 64 + 16 * g, row * 144 + 96 + 16 * g);
    }

    float rmax[8], rsum[8];
    v8f accO[4] = {};
    #pragma unroll
    for (int f = 0; f < 8; ++f) { rmax[f] = -1e30f; rsum[f] = 0.0f; }

    f16_t* myPs = Ps + wave * 16 * 72;
    const int jtMax = 2 * qt + 1;       // kv tiles of 64 covering q rows

    for (int jt = 0; jt <= jtMax; ++jt) {
        const int kvBase = jt * 64;
        __syncthreads();
        // ---- stage K tile: 64 x 64 halves = 512 x 16B ----
        #pragma unroll
        for (int i = 0; i < 2; ++i) {
            int u = tid + i * 256;
            int row = u >> 3, c8 = u & 7;
            v4u d = *reinterpret_cast<const v4u*>(Kt + headOff + (size_t)(kvBase + row) * D_K + c8 * 8);
            *reinterpret_cast<v4u*>(reinterpret_cast<char*>(Ks) + row * 144 + c8 * 16) = d;
        }
        // ---- stage V tile transposed ----
        #pragma unroll
        for (int e = 0; e < 16; ++e) {
            int linear = tid + e * 256;
            int kv = linear >> 6, d = linear & 63;
            Vts[d * 72 + kv] = V[headOff + (size_t)(kvBase + kv) * D_K + d];
        }
        __syncthreads();

        // Waves entirely above the diagonal contribute nothing for this tile.
        const bool active = (kvBase <= waveRow0 + 15);

        if (active) {
            // ---- S = Q K^T ----
            v8f accS[4] = {};
            #pragma unroll
            for (int kk = 0; kk < 2; ++kk) {
                #pragma unroll
                for (int tj = 0; tj < 4; ++tj) {
                    int nrow = tj * 16 + r;
                    v16h bk = load_frag(Ks, nrow * 144 + kk * 64 + 16 * g,
                                            nrow * 144 + kk * 64 + 32 + 16 * g);
                    accS[tj] = __builtin_amdgcn_wmma_f32_16x16x32_f16(
                        false, aq[kk], false, bk, (short)0, accS[tj], false, false);
                }
            }

            // ---- online softmax ----
            const bool needMask = (kvBase + 63 > waveRow0);
            float pvals[4][8];
            #pragma unroll
            for (int f = 0; f < 8; ++f) {
                int mGlob = waveRow0 + 8 * g + f;
                float mval = -1e30f;
                #pragma unroll
                for (int tj = 0; tj < 4; ++tj) {
                    float sv = accS[tj][f] * 0.125f;   // 1/sqrt(64)
                    if (needMask) {
                        int nGlob = kvBase + tj * 16 + r;
                        if (nGlob > mGlob) sv = -1e30f;
                    }
                    pvals[tj][f] = sv;
                    mval = fmaxf(mval, sv);
                }
                #pragma unroll
                for (int mk = 1; mk < 16; mk <<= 1)
                    mval = fmaxf(mval, __shfl_xor(mval, mk, 32));
                float nmax  = fmaxf(rmax[f], mval);
                float alpha = __expf(rmax[f] - nmax);
                rmax[f] = nmax;
                float psum = 0.0f;
                #pragma unroll
                for (int tj = 0; tj < 4; ++tj) {
                    float p = __expf(pvals[tj][f] - nmax);
                    pvals[tj][f] = p;
                    psum += p;
                }
                #pragma unroll
                for (int mk = 1; mk < 16; mk <<= 1)
                    psum += __shfl_xor(psum, mk, 32);
                rsum[f] = rsum[f] * alpha + psum;
                #pragma unroll
                for (int tj = 0; tj < 4; ++tj) accO[tj][f] *= alpha;
                // stash P (C-layout) into LDS for A-layout reload
                #pragma unroll
                for (int tj = 0; tj < 4; ++tj)
                    myPs[(8 * g + f) * 72 + tj * 16 + r] = (f16_t)pvals[tj][f];
            }
        }
        __syncthreads();   // uniform; also orders per-wave Ps stores vs reloads

        if (active) {
            // ---- O += P V ----
            #pragma unroll
            for (int kk = 0; kk < 2; ++kk) {
                v16h ap = load_frag(myPs, r * 144 + kk * 64 + 16 * g,
                                          r * 144 + kk * 64 + 32 + 16 * g);
                #pragma unroll
                for (int tj = 0; tj < 4; ++tj) {
                    int nrow = tj * 16 + r;
                    v16h bv = load_frag(Vts, nrow * 144 + kk * 64 + 16 * g,
                                             nrow * 144 + kk * 64 + 32 + 16 * g);
                    accO[tj] = __builtin_amdgcn_wmma_f32_16x16x32_f16(
                        false, ap, false, bv, (short)0, accO[tj], false, false);
                }
            }
        }
    }

    // ---- normalize + write [B,S,1024] f16 ----
    #pragma unroll
    for (int f = 0; f < 8; ++f) {
        float inv = 1.0f / rsum[f];
        int sIdx = waveRow0 + 8 * g + f;
        #pragma unroll
        for (int tj = 0; tj < 4; ++tj) {
            int dcol = tj * 16 + r;
            outH[((size_t)b * SEQ + sIdx) * D_MODEL + h * D_K + dcol] =
                (f16_t)(accO[tj][f] * inv);
        }
    }
}

// ---------------------------------------------------------------------------
// Host-side orchestration
// ---------------------------------------------------------------------------
extern "C" void kernel_launch(void* const* d_in, const int* in_sizes, int n_in,
                              void* d_out, int out_size, void* d_ws, size_t ws_size,
                              hipStream_t stream) {
    const float* x      = (const float*)d_in[0];
    const int*   tokpos = (const int*)  d_in[1];
    const float* Qw     = (const float*)d_in[2];
    const float* Kw     = (const float*)d_in[3];
    const float* Vw     = (const float*)d_in[4];
    const float* Ow     = (const float*)d_in[5];
    float* out = (float*)d_out;

    char* ws = (char*)d_ws;
    const size_t xElems = (size_t)M_TOTAL * D_MODEL;       // 8M
    const size_t wElems = (size_t)D_MODEL * D_MODEL;       // 1M
    f16_t* xh    = (f16_t*)ws;  ws += xElems * 2;
    f16_t* Qwh   = (f16_t*)ws;  ws += wElems * 2;
    f16_t* Kwh   = (f16_t*)ws;  ws += wElems * 2;
    f16_t* Vwh   = (f16_t*)ws;  ws += wElems * 2;
    f16_t* Owh   = (f16_t*)ws;  ws += wElems * 2;
    f16_t* Qh    = (f16_t*)ws;  ws += xElems * 2;          // [B,H,S,64]
    f16_t* Kh    = (f16_t*)ws;  ws += xElems * 2;
    f16_t* Vh    = (f16_t*)ws;  ws += xElems * 2;
    f16_t* attnh = (f16_t*)ws;  ws += xElems * 2;          // [B,S,1024]

    // 1) convert to f16
    {
        int n4 = (int)(xElems / 4);
        cvt_f32_f16<<<(n4 + 255) / 256, 256, 0, stream>>>(x, xh, n4);
        int w4 = (int)(wElems / 4);
        cvt_f32_f16<<<(w4 + 255) / 256, 256, 0, stream>>>(Qw, Qwh, w4);
        cvt_f32_f16<<<(w4 + 255) / 256, 256, 0, stream>>>(Kw, Kwh, w4);
        cvt_f32_f16<<<(w4 + 255) / 256, 256, 0, stream>>>(Vw, Vwh, w4);
        cvt_f32_f16<<<(w4 + 255) / 256, 256, 0, stream>>>(Ow, Owh, w4);
    }

    // 2) QKV projections (+ fused RoPE on Q,K), scatter to [B,H,S,64]
    dim3 ggrid(D_MODEL / 128, M_TOTAL / 128);   // (8, 64)
    gemm_nt_kernel<<<ggrid, 256, 0, stream>>>(xh, Qwh, tokpos, Qh, nullptr,
                                              D_MODEL, D_MODEL, /*rope=*/1, /*mode=*/1);
    gemm_nt_kernel<<<ggrid, 256, 0, stream>>>(xh, Kwh, tokpos, Kh, nullptr,
                                              D_MODEL, D_MODEL, /*rope=*/1, /*mode=*/1);
    gemm_nt_kernel<<<ggrid, 256, 0, stream>>>(xh, Vwh, tokpos, Vh, nullptr,
                                              D_MODEL, D_MODEL, /*rope=*/0, /*mode=*/1);

    // 3) causal flash attention -> [B,S,1024] f16
    attn_kernel<<<BATCH * NUM_HEADS * (SEQ / 128), 256, 0, stream>>>(Qh, Kh, Vh, attnh);

    // 4) output projection -> d_out f32
    gemm_nt_kernel<<<ggrid, 256, 0, stream>>>(attnh, Owh, tokpos, nullptr, out,
                                              D_MODEL, D_MODEL, /*rope=*/0, /*mode=*/2);
}
